// MultiCNNTransformerBlock_vis_15917148799145
// MI455X (gfx1250) — matrix operand order repaired and checked by reading, not compiled, not run on hardware
//
#include <hip/hip_runtime.h>

typedef _Float16 f16;
typedef __attribute__((ext_vector_type(8)))  _Float16 v8h;
typedef __attribute__((ext_vector_type(16))) _Float16 v16h;
typedef __attribute__((ext_vector_type(8)))  float    v8f;

#define DEVFN static __device__ __forceinline__

static constexpr int DMODEL = 256;
static constexpr int LTOK = 1024;       // H*W
static constexpr int MROWS = 8 * LTOK;  // 8192

DEVFN v8f wmma32(v16h a, v16h b, v8f c) {
  return __builtin_amdgcn_wmma_f32_16x16x32_f16(false, a, false, b, (short)0, c,
                                                false, false);
}

// A fragment (16x32 f16, row-major, stride ld halves)
DEVFN v16h frag_a(const f16* t, int ld) {
  int lane = threadIdx.x & 31;
  const f16* p = t + (lane & 15) * ld + ((lane & 16) ? 8 : 0);
  v8h lo = *(const v8h*)p;
  v8h hi = *(const v8h*)(p + 16);
  v16h a;
#pragma unroll
  for (int i = 0; i < 8; ++i) { a[i] = lo[i]; a[8 + i] = hi[i]; }
  return a;
}

// B fragment (32x16 f16 tile of row-major K x N buffer, stride ld halves)
DEVFN v16h frag_b(const f16* t, int ld) {
  int lane = threadIdx.x & 31;
  const f16* p = t + ((lane & 16) ? 8 : 0) * ld + (lane & 15);
  v16h b;
#pragma unroll
  for (int i = 0; i < 8; ++i) { b[i] = p[i * ld]; b[8 + i] = p[(16 + i) * ld]; }
  return b;
}

DEVFN float elup1(float x) { return x > 0.f ? x + 1.f : __expf(x); }

// async direct-to-LDS 16-byte load (ASYNCcnt); lds/byte offsets, SGPR base
DEVFN void async_b128(unsigned ldsOff, unsigned byteOff, const void* base) {
  asm volatile("global_load_async_to_lds_b128 %0, %1, %2"
               :
               : "v"(ldsOff), "v"(byteOff), "s"(base)
               : "memory");
}
DEVFN void wait_async0() {
  asm volatile("s_wait_asynccnt 0x0" ::: "memory");
}

// ---------------------------------------------------------------- converts
__global__ __launch_bounds__(256) void nchw_to_cl(const float* __restrict__ in,
                                                  f16* __restrict__ out, int ldo) {
  int idx = blockIdx.x * 256 + threadIdx.x;
  int m = idx >> 8, c = idx & 255;
  int b = m >> 10, l = m & 1023;
  out[(size_t)m * ldo + c] = (f16)in[((size_t)(b * DMODEL + c)) * LTOK + l];
}

__global__ __launch_bounds__(256) void cvt_f16(const float* __restrict__ in,
                                               f16* __restrict__ out, int n) {
  int i = blockIdx.x * 256 + threadIdx.x;
  if (i < n) out[i] = (f16)in[i];
}

// Repack conv weight [CO][256][k][k] f32 -> [tap][ci][CO] f16 (tap = kh*k+kw)
__global__ __launch_bounds__(256) void prep_conv_w(const float* __restrict__ w,
                                                   f16* __restrict__ dst, int CO,
                                                   int ksz) {
  int idx = blockIdx.x * 256 + threadIdx.x;
  int total = CO * 256 * ksz * ksz;
  if (idx >= total) return;
  int co = idx % CO;
  int rest = idx / CO;
  int ci = rest % 256;
  int t = rest / 256;
  dst[idx] = (f16)w[((size_t)(co * 256 + ci)) * ksz * ksz + t];
}

// --------------------------------------- unified implicit-GEMM conv / GEMM
// 64x64 macro-tile, 4 waves x (2x2) accumulators, K stage = 64.
// Double-buffered LDS staging via GLOBAL_LOAD_ASYNC_TO_LDS_B128.
// MODE: 0 = f16 out[m*ldc+colOff+n]
//       1 = f16 kT  out[((b*4+h)*64+d)*1024 + s]
//       2 = f32 out
//       3 = relu + f16 out
template <int KSZ, int PAD, int CI, int MODE>
__global__ __launch_bounds__(128) void mmconv_wmma(
    const f16* __restrict__ src, int lda, const f16* __restrict__ wt, int ldb,
    void* __restrict__ out, int colOff, int ldc) {
  extern __shared__ __align__(32) f16 sm[];   // 2 x (64x64 A + 64x64 B)
  constexpr int TAPS = KSZ * KSZ;
  constexpr int CHUNKS = CI >> 6;             // K stage = 64
  const unsigned ldsBase = __builtin_amdgcn_groupstaticsize();
  const int m0 = blockIdx.x * 64;
  const int n0 = blockIdx.y * 64;
  const int bb_ = m0 >> 10;
  const int y0 = (m0 & 1023) >> 5;            // image rows y0, y0+1
  const int tid = threadIdx.x, wv = tid >> 5, lane = tid & 31;
  const int mw = wv & 1, nw = wv >> 1;
  v8f acc[2][2] = {};

  auto tapOK = [&](int t) -> bool {
    int dy = t / KSZ - PAD;
    return (y0 + 1 + dy >= 0) && (y0 + dy <= 31);
  };
  auto issue = [&](int t, int kc, int bufsel) {
    int dy = t / KSZ - PAD, dx = t % KSZ - PAD;
    int ci0 = kc * 64;
    int hbuf = bufsel * 8192;                 // halves
#pragma unroll
    for (int j = 0; j < 4; ++j) {
      int i = tid + j * 128;
      int row = i >> 3, c8 = (i & 7) * 8;
      {  // A tile: 64 pixels x 64 channels
        int yy = y0 + (row >> 5) + dy;
        int xx = (row & 31) + dx;
        int hidx = hbuf + row * 64 + c8;
        if (yy >= 0 && yy < 32 && xx >= 0 && xx < 32) {
          unsigned go =
              (unsigned)((((bb_ * 32 + yy) * 32 + xx) * lda + ci0 + c8) * 2);
          async_b128(ldsBase + hidx * 2, go, src);
        } else {
          v8h z = {};
          *(v8h*)&sm[hidx] = z;
        }
      }
      {  // B tile: 64 K-rows x 64 N
        int hidx = hbuf + 4096 + row * 64 + c8;
        unsigned go = (unsigned)(((t * CI + ci0 + row) * ldb + n0 + c8) * 2);
        async_b128(ldsBase + hidx * 2, go, wt);
      }
    }
  };
  auto compute = [&](int bufsel) {
    const f16* base = sm + bufsel * 8192;
#pragma unroll
    for (int kh = 0; kh < 2; ++kh) {
      v16h a0 = frag_a(base + (mw * 32) * 64 + kh * 32, 64);
      v16h a1 = frag_a(base + (mw * 32 + 16) * 64 + kh * 32, 64);
      const f16* bp = base + 4096 + (kh * 32) * 64 + nw * 32;
      v16h b0 = frag_b(bp, 64);
      v16h b1 = frag_b(bp + 16, 64);
      acc[0][0] = wmma32(a0, b0, acc[0][0]);
      acc[0][1] = wmma32(a0, b1, acc[0][1]);
      acc[1][0] = wmma32(a1, b0, acc[1][0]);
      acc[1][1] = wmma32(a1, b1, acc[1][1]);
    }
  };

  int t = 0;
  while (t < TAPS && !tapOK(t)) ++t;
  int kc = 0, buf = 0;
  issue(t, kc, 0);
  while (t < TAPS) {
    wait_async0();
    __syncthreads();
    int tn = t, kn = kc + 1;
    if (kn == CHUNKS) {
      kn = 0;
      ++tn;
      while (tn < TAPS && !tapOK(tn)) ++tn;
    }
    if (tn < TAPS) issue(tn, kn, buf ^ 1);
    compute(buf);
    __syncthreads();
    t = tn;
    kc = kn;
    buf ^= 1;
  }

  int lane15 = lane & 15, rb = (lane >> 4) << 3;
#pragma unroll
  for (int im = 0; im < 2; ++im)
#pragma unroll
    for (int in = 0; in < 2; ++in)
#pragma unroll
      for (int r = 0; r < 8; ++r) {
        int gm = m0 + mw * 32 + im * 16 + rb + r;
        int gn = n0 + nw * 32 + in * 16 + lane15;
        float v = acc[im][in][r];
        if constexpr (MODE == 1) {
          int c = colOff + gn, h = c >> 6, d = c & 63;
          ((f16*)out)[(size_t)(((gm >> 10) * 4 + h) * 64 + d) * 1024 +
                      (gm & 1023)] = (f16)v;
        } else if constexpr (MODE == 2) {
          ((float*)out)[(size_t)gm * ldc + colOff + gn] = v;
        } else {
          if constexpr (MODE == 3) v = v > 0.f ? v : 0.f;
          ((f16*)out)[(size_t)gm * ldc + colOff + gn] = (f16)v;
        }
      }
}

// ------------------------------------------- fused qk GEMM + softmax -> vis
__global__ __launch_bounds__(128) void attn_vis_kernel(
    const f16* __restrict__ q_cl, const f16* __restrict__ k_t,
    float* __restrict__ vis) {
  __shared__ float P[16][1024];                      // 64 KB
  int bh = blockIdx.x >> 6;
  int l0 = (blockIdx.x & 63) * 16;
  int b = bh >> 2, h = bh & 3;
  int tid = threadIdx.x, wv = tid >> 5, lane = tid & 31;
  v16h aq0, aq1;
  {
    int r = lane & 15, c0 = (lane & 16) ? 8 : 0;
    const f16* qp = q_cl + (size_t)(b * 1024 + l0 + r) * 256 + h * 64 + c0;
    v8h a = *(const v8h*)qp, bv = *(const v8h*)(qp + 16);
    v8h cv = *(const v8h*)(qp + 32), dv = *(const v8h*)(qp + 48);
#pragma unroll
    for (int i = 0; i < 8; ++i) {
      aq0[i] = a[i]; aq0[8 + i] = bv[i];
      aq1[i] = cv[i]; aq1[8 + i] = dv[i];
    }
  }
  const f16* kp = k_t + (size_t)bh * 64 * 1024;
  for (int nt = 0; nt < 16; ++nt) {
    int s0 = wv * 256 + nt * 16;
    v8f acc = {};
    acc = wmma32(aq0, frag_b(kp + s0, 1024), acc);
    acc = wmma32(aq1, frag_b(kp + (size_t)32 * 1024 + s0, 1024), acc);
    int col = s0 + (lane & 15);
    int rb = (lane >> 4) << 3;
#pragma unroll
    for (int r = 0; r < 8; ++r) P[rb + r][col] = acc[r] * 0.125f;  // DH^-0.5
  }
  __syncthreads();
  int row = tid >> 3, sub = tid & 7;
  float* pr = P[row] + sub * 128;
  float mx = -3.0e38f;
  for (int i = 0; i < 128; ++i) mx = fmaxf(mx, pr[i]);
#pragma unroll
  for (int k = 1; k < 8; k <<= 1) mx = fmaxf(mx, __shfl_xor(mx, k));
  float sum = 0.f;
  for (int i = 0; i < 128; ++i) { float e = __expf(pr[i] - mx); pr[i] = e; sum += e; }
#pragma unroll
  for (int k = 1; k < 8; k <<= 1) sum += __shfl_xor(sum, k);
  float inv = 1.f / sum;
  float* vp = vis + (size_t)(bh * 1024 + l0 + row) * 1024 + sub * 128;
  for (int i = 0; i < 128; ++i) vp[i] = pr[i] * inv;
}

// ------------------------------------------------------- linear attention
__global__ __launch_bounds__(128) void kfsum_kernel(const f16* __restrict__ k_t,
                                                    float* __restrict__ kfs) {
  int bd = blockIdx.x;                               // bh*64 + d
  const f16* p = k_t + (size_t)bd * 1024;
  float s = 0.f;
  for (int i = threadIdx.x; i < 1024; i += 128) s += elup1((float)p[i]);
#pragma unroll
  for (int k = 16; k >= 1; k >>= 1) s += __shfl_xor(s, k);
  __shared__ float r[4];
  if ((threadIdx.x & 31) == 0) r[threadIdx.x >> 5] = s;
  __syncthreads();
  if (threadIdx.x == 0) kfs[bd] = r[0] + r[1] + r[2] + r[3];
}

__global__ __launch_bounds__(128) void kv_kernel(const f16* __restrict__ k_t,
                                                 const f16* __restrict__ v_cl,
                                                 f16* __restrict__ kv) {
  int bh = blockIdx.x, b = bh >> 2, h = bh & 3;
  int tid = threadIdx.x, wv = tid >> 5, lane = tid & 31;
  int r = lane & 15, c0 = (lane & 16) ? 8 : 0;
  v8f acc[4] = {};
  const f16* kp = k_t + (size_t)bh * 64 * 1024;
  const f16* vp = v_cl + (size_t)b * 1024 * 256 + h * 64;
  for (int kc = 0; kc < 32; ++kc) {
    v16h af;
    {
      const f16* ap = kp + (size_t)(wv * 16 + r) * 1024 + kc * 32 + c0;
#pragma unroll
      for (int i = 0; i < 8; ++i) {
        af[i] = (f16)elup1((float)ap[i]);
        af[8 + i] = (f16)elup1((float)ap[i + 16]);
      }
    }
    const f16* bp = vp + (size_t)(kc * 32) * 256;
#pragma unroll
    for (int n = 0; n < 4; ++n) {
      v16h bf;
#pragma unroll
      for (int j = 0; j < 16; ++j)
        bf[j] = (f16)((float)bp[(size_t)(c0 + (j & 7) + ((j >> 3) << 4)) * 256 + n * 16 + r] * (1.f / 1024.f));
      acc[n] = wmma32(af, bf, acc[n]);
    }
  }
  int col = lane & 15, rb = (lane >> 4) << 3;
  f16* outp = kv + (size_t)bh * 4096;
#pragma unroll
  for (int n = 0; n < 4; ++n)
#pragma unroll
    for (int rr = 0; rr < 8; ++rr)
      outp[(size_t)(wv * 16 + rb + rr) * 64 + n * 16 + col] = (f16)acc[n][rr];
}

__global__ __launch_bounds__(128) void msg_kernel(const f16* __restrict__ q_cl,
                                                  const f16* __restrict__ kv,
                                                  const float* __restrict__ kfsum,
                                                  f16* __restrict__ msg) {
  __shared__ float Zs[32][4];
  __shared__ float kfs[4][64];
  int b = blockIdx.x >> 5;
  int l0 = (blockIdx.x & 31) * 32;
  int tid = threadIdx.x, wv = tid >> 5, lane = tid & 31;
  for (int i = tid; i < 256; i += 128)
    kfs[i >> 6][i & 63] = kfsum[(b * 4 + (i >> 6)) * 64 + (i & 63)];
  __syncthreads();
  {
    int l = tid >> 2, h = tid & 3;
    const f16* qp = q_cl + (size_t)(b * 1024 + l0 + l) * 256 + h * 64;
    float a = 0.f;
    for (int d = 0; d < 64; ++d) a += elup1((float)qp[d]) * kfs[h][d];
    Zs[l][h] = 1024.f / (a + 1e-6f);                 // fold * v_len
  }
  __syncthreads();
  int r = lane & 15, c0 = (lane & 16) ? 8 : 0;
  int mh = wv & 1, nb = (wv >> 1) * 32;
  for (int h = 0; h < 4; ++h) {
    v8f acc[2] = {};
#pragma unroll
    for (int kc = 0; kc < 2; ++kc) {
      v16h af;
      const f16* qp = q_cl + (size_t)(b * 1024 + l0 + mh * 16 + r) * 256 +
                      h * 64 + kc * 32 + c0;
#pragma unroll
      for (int i = 0; i < 8; ++i) {
        af[i] = (f16)elup1((float)qp[i]);
        af[8 + i] = (f16)elup1((float)qp[i + 16]);
      }
      const f16* bp = kv + (size_t)(b * 4 + h) * 4096 + (size_t)(kc * 32) * 64 + nb;
      acc[0] = wmma32(af, frag_b(bp, 64), acc[0]);
      acc[1] = wmma32(af, frag_b(bp + 16, 64), acc[1]);
    }
    int col = lane & 15, rb = (lane >> 4) << 3;
#pragma unroll
    for (int n = 0; n < 2; ++n)
#pragma unroll
      for (int rr = 0; rr < 8; ++rr) {
        int ml = mh * 16 + rb + rr;
        msg[(size_t)(b * 1024 + l0 + ml) * 256 + h * 64 + nb + n * 16 + col] =
            (f16)(acc[n][rr] * Zs[ml][h]);
      }
  }
}

// ----------------------------------------------------------- layer norm
__global__ __launch_bounds__(256) void ln_kernel(
    const float* __restrict__ in, const float* __restrict__ g,
    const float* __restrict__ bb, int mode, f16* __restrict__ outH, int ldo,
    int colOff, float* __restrict__ outF, const float* __restrict__ xres) {
  __shared__ float sm[256];
  int m = blockIdx.x, c = threadIdx.x;
  float v = in[(size_t)m * 256 + c];
  sm[c] = v;
  __syncthreads();
  for (int s = 128; s > 0; s >>= 1) {
    if (c < s) sm[c] += sm[c + s];
    __syncthreads();
  }
  float mean = sm[0] * (1.f / 256.f);
  __syncthreads();
  float d = v - mean;
  sm[c] = d * d;
  __syncthreads();
  for (int s = 128; s > 0; s >>= 1) {
    if (c < s) sm[c] += sm[c + s];
    __syncthreads();
  }
  float var = sm[0] * (1.f / 256.f);
  float y = d * rsqrtf(var + 1e-5f) * g[c] + bb[c];
  if (mode == 0) {
    outH[(size_t)m * ldo + colOff + c] = (f16)y;
  } else {
    int b = m >> 10, l = m & 1023;
    size_t idx = (size_t)(b * 256 + c) * 1024 + l;
    outF[idx] = xres[idx] + y;
  }
}

// ================================================================== launch
extern "C" void kernel_launch(void* const* d_in, const int* in_sizes, int n_in,
                              void* d_out, int out_size, void* d_ws,
                              size_t ws_size, hipStream_t stream) {
  (void)in_sizes; (void)n_in; (void)out_size; (void)ws_size;
  const float* x    = (const float*)d_in[0];
  const float* src  = (const float*)d_in[1];
  const float* Wq   = (const float*)d_in[2];
  const float* Wk[4]= {(const float*)d_in[3], (const float*)d_in[4],
                       (const float*)d_in[5], (const float*)d_in[6]};
  const float* Wv[4]= {(const float*)d_in[7], (const float*)d_in[8],
                       (const float*)d_in[9], (const float*)d_in[10]};
  const float* Wm   = (const float*)d_in[11];
  const float* ln1g = (const float*)d_in[12];
  const float* ln1b = (const float*)d_in[13];
  const float* ln2g = (const float*)d_in[14];
  const float* ln2b = (const float*)d_in[15];
  const float* W1   = (const float*)d_in[16];
  const float* W2   = (const float*)d_in[17];
  float* outp = (float*)d_out;
  float* vis  = outp + (size_t)MROWS * 256;          // out then vis

  char* w = (char*)d_ws;
  size_t off = 0;
  auto nxt = [&](size_t bytes) -> char* {
    char* p = w + off;
    off += (bytes + 255) & ~(size_t)255;
    return p;
  };
  f16*   cat   = (f16*)nxt((size_t)MROWS * 512 * 2); // [xf | msg_ln]
  f16*   srcCL = (f16*)nxt((size_t)MROWS * 256 * 2);
  f16*   qcl   = (f16*)nxt((size_t)MROWS * 256 * 2);
  f16*   kT    = (f16*)nxt((size_t)32 * 64 * 1024 * 2);
  f16*   vcl   = (f16*)nxt((size_t)MROWS * 256 * 2);
  f16*   wqT   = (f16*)nxt((size_t)9 * 256 * 256 * 2);
  f16*   wkT   = (f16*)nxt((size_t)84 * 256 * 64 * 2);
  f16*   wvT   = (f16*)nxt((size_t)84 * 256 * 64 * 2);
  f16*   wmH   = (f16*)nxt((size_t)256 * 256 * 2);
  f16*   w1H   = (f16*)nxt((size_t)512 * 512 * 2);
  f16*   w2H   = (f16*)nxt((size_t)512 * 256 * 2);
  f16*   kvH   = (f16*)nxt((size_t)32 * 64 * 64 * 2);
  float* kfs   = (float*)nxt((size_t)32 * 64 * 4);
  f16*   msgH  = (f16*)nxt((size_t)MROWS * 256 * 2);
  float* mrg   = (float*)nxt((size_t)MROWS * 256 * 4);
  f16*   hidH  = (f16*)nxt((size_t)MROWS * 512 * 2);
  float* ffF   = (float*)nxt((size_t)MROWS * 256 * 4);

  nchw_to_cl<<<MROWS, 256, 0, stream>>>(x, cat, 512);
  nchw_to_cl<<<MROWS, 256, 0, stream>>>(src, srcCL, 256);

  prep_conv_w<<<(589824 + 255) / 256, 256, 0, stream>>>(Wq, wqT, 256, 3);
  const int ks[4] = {1, 3, 5, 7};
  const int tof[4] = {0, 1, 10, 35};
  for (int s = 0; s < 4; ++s) {
    int n = 64 * 256 * ks[s] * ks[s];
    prep_conv_w<<<(n + 255) / 256, 256, 0, stream>>>(
        Wk[s], wkT + (size_t)tof[s] * 256 * 64, 64, ks[s]);
    prep_conv_w<<<(n + 255) / 256, 256, 0, stream>>>(
        Wv[s], wvT + (size_t)tof[s] * 256 * 64, 64, ks[s]);
  }
  cvt_f16<<<256, 256, 0, stream>>>(Wm, wmH, 65536);
  cvt_f16<<<1024, 256, 0, stream>>>(W1, w1H, 262144);
  cvt_f16<<<512, 256, 0, stream>>>(W2, w2H, 131072);

  const size_t SMEM = 32768;  // 2 x (8KB A + 8KB B)

  // q = conv3x3(x) -> qcl
  mmconv_wmma<3, 1, 256, 0><<<dim3(128, 4), 128, SMEM, stream>>>(
      cat, 512, wqT, 256, qcl, 0, 256);
  // k (transposed layout) / v multi-scale convs
  mmconv_wmma<1, 0, 256, 1><<<dim3(128, 1), 128, SMEM, stream>>>(
      srcCL, 256, wkT + (size_t)tof[0] * 16384, 64, kT, 0, 0);
  mmconv_wmma<3, 1, 256, 1><<<dim3(128, 1), 128, SMEM, stream>>>(
      srcCL, 256, wkT + (size_t)tof[1] * 16384, 64, kT, 64, 0);
  mmconv_wmma<5, 2, 256, 1><<<dim3(128, 1), 128, SMEM, stream>>>(
      srcCL, 256, wkT + (size_t)tof[2] * 16384, 64, kT, 128, 0);
  mmconv_wmma<7, 3, 256, 1><<<dim3(128, 1), 128, SMEM, stream>>>(
      srcCL, 256, wkT + (size_t)tof[3] * 16384, 64, kT, 192, 0);
  mmconv_wmma<1, 0, 256, 0><<<dim3(128, 1), 128, SMEM, stream>>>(
      srcCL, 256, wvT + (size_t)tof[0] * 16384, 64, vcl, 0, 256);
  mmconv_wmma<3, 1, 256, 0><<<dim3(128, 1), 128, SMEM, stream>>>(
      srcCL, 256, wvT + (size_t)tof[1] * 16384, 64, vcl, 64, 256);
  mmconv_wmma<5, 2, 256, 0><<<dim3(128, 1), 128, SMEM, stream>>>(
      srcCL, 256, wvT + (size_t)tof[2] * 16384, 64, vcl, 128, 256);
  mmconv_wmma<7, 3, 256, 0><<<dim3(128, 1), 128, SMEM, stream>>>(
      srcCL, 256, wvT + (size_t)tof[3] * 16384, 64, vcl, 192, 256);

  attn_vis_kernel<<<2048, 128, 0, stream>>>(qcl, kT, vis);

  kfsum_kernel<<<2048, 128, 0, stream>>>(kT, kfs);
  kv_kernel<<<32, 128, 0, stream>>>(kT, vcl, kvH);
  msg_kernel<<<256, 128, 0, stream>>>(qcl, kvH, kfs, msgH);

  // merge (f32) -> LN1 -> concat cols 256..511
  mmconv_wmma<1, 0, 256, 2><<<dim3(128, 4), 128, SMEM, stream>>>(
      msgH, 256, wmH, 256, mrg, 0, 256);
  ln_kernel<<<MROWS, 256, 0, stream>>>(mrg, ln1g, ln1b, 0, cat, 512, 256,
                                       nullptr, nullptr);

  // FF
  mmconv_wmma<1, 0, 512, 3><<<dim3(128, 8), 128, SMEM, stream>>>(
      cat, 512, w1H, 512, hidH, 0, 512);
  mmconv_wmma<1, 0, 512, 2><<<dim3(128, 4), 128, SMEM, stream>>>(
      hidH, 512, w2H, 256, ffF, 0, 256);
  ln_kernel<<<MROWS, 256, 0, stream>>>(ffF, ln2g, ln2b, 1, nullptr, 0, 0,
                                       outp, x);
}
